// WaveResNetMamba_6597069767167
// MI455X (gfx1250) — matrix operand order repaired and checked by reading, not compiled, not run on hardware
//
#include <hip/hip_runtime.h>

// ---------------------------------------------------------------------------
// Types for CDNA5 WMMA (wave32, v_wmma_f32_16x16x32_bf16)
// ---------------------------------------------------------------------------
typedef __attribute__((ext_vector_type(16))) __bf16 v16bf;
typedef __attribute__((ext_vector_type(8)))  __bf16 v8bf;
typedef __attribute__((ext_vector_type(8)))  float  v8f;

// ---------------------------------------------------------------------------
// Model dimensions (compile-time, from the reference)
// ---------------------------------------------------------------------------
#define B_SZ     8
#define T_IN     16384
#define L_DWT    8195      // (16384+7)//2
#define L_STEM   4098      // conv k7 s2 p3
#define L_POOL   2049      // maxpool k3 s2 p1
#define L_TOK    1025      // after l2 stride-2
#define DM       128       // d_model
#define DI       256       // d_inner
#define DS       16        // d_state
#define DTR      8         // dt_rank
#define M_ROWS   (B_SZ * L_TOK)   // 8200 token rows
#define MP_ROWS  8208             // 513*16 (padded to WMMA M tiles)
#define DBL_LD   64               // x_proj output padded: 40 -> 64 (N tile mult of 64)
#define XZ_LD    512

#define GE(n) ((unsigned)(((n) + 255) / 256))

// DWT filters (already reversed, i.e. cross-correlation taps = DEC[::-1])
__device__ __constant__ float c_klo[8] = {
   0.23037781330885523f,  0.7148465705525415f,  0.6308807679295904f,
  -0.02798376941698385f, -0.18703481171888114f, 0.030841381835986965f,
   0.032883011666982945f, -0.010597401784997278f };
__device__ __constant__ float c_khi[8] = {
  -0.010597401784997278f, 0.032883011666982945f, 0.030841381835986965f,
   0.18703481171888114f, -0.02798376941698385f, -0.6308807679295904f,
   0.7148465705525415f,  -0.23037781330885523f };

__device__ inline __bf16 f2bf(float f) {
  unsigned u = __builtin_bit_cast(unsigned, f);
  u = (u + 0x7FFFu + ((u >> 16) & 1u)) >> 16;
  unsigned short s = (unsigned short)u;
  return __builtin_bit_cast(__bf16, s);
}
__device__ inline float sigmoidf_(float x) { return 1.f / (1.f + __expf(-x)); }
__device__ inline float gelu_(float x) { return 0.5f * x * (1.f + erff(x * 0.70710678118654752f)); }

// ---------------------------------------------------------------------------
// 1. DWT with reflect padding (pad 6 each side), stride 2, 8 taps
// ---------------------------------------------------------------------------
__global__ void k_dwt(const float* __restrict__ x, float* __restrict__ yl,
                      float* __restrict__ yh) {
  int idx = blockIdx.x * blockDim.x + threadIdx.x;
  if (idx >= B_SZ * L_DWT) return;
  int o = idx % L_DWT, b = idx / L_DWT;
  const float* xb = x + (size_t)b * T_IN;
  float sl = 0.f, sh = 0.f;
#pragma unroll
  for (int i = 0; i < 8; ++i) {
    int j = 2 * o + i - 6;
    if (j < 0) j = -j;
    else if (j >= T_IN) j = 2 * T_IN - 2 - j;
    float v = xb[j];
    sl += v * c_klo[i];
    sh += v * c_khi[i];
  }
  yl[idx] = sl; yh[idx] = sh;
}

// ---------------------------------------------------------------------------
// 2. Generic NCH conv1d (cross-correlation), zero padding
// ---------------------------------------------------------------------------
__global__ void k_conv1d(const float* __restrict__ in, const float* __restrict__ w,
                         float* __restrict__ out, int Cin, int Cout,
                         int Lin, int Lout, int K, int S, int P) {
  int idx = blockIdx.x * blockDim.x + threadIdx.x;
  int tot = B_SZ * Cout * Lout;
  if (idx >= tot) return;
  int o  = idx % Lout;
  int co = (idx / Lout) % Cout;
  int b  = idx / (Lout * Cout);
  float acc = 0.f;
  for (int ci = 0; ci < Cin; ++ci) {
    const float* ip = in + ((size_t)b * Cin + ci) * Lin;
    const float* wp = w + ((size_t)co * Cin + ci) * K;
    for (int i = 0; i < K; ++i) {
      int li = o * S + i - P;
      if (li >= 0 && li < Lin) acc += wp[i] * ip[li];
    }
  }
  out[idx] = acc;
}

// ---------------------------------------------------------------------------
// 3. BatchNorm statistics (mean over (B,L) per channel, biased var)
// ---------------------------------------------------------------------------
__global__ void k_bn_stats(const float* __restrict__ x, float* __restrict__ mean,
                           float* __restrict__ istd, int C, int Lc) {
  int c = blockIdx.x;
  int tid = threadIdx.x;
  __shared__ float ss[256], sq[256];
  float s = 0.f, q = 0.f;
  int tot = B_SZ * Lc;
  for (int i = tid; i < tot; i += 256) {
    int b = i / Lc, l = i % Lc;
    float v = x[((size_t)b * C + c) * Lc + l];
    s += v; q += v * v;
  }
  ss[tid] = s; sq[tid] = q;
  __syncthreads();
  for (int st = 128; st > 0; st >>= 1) {
    if (tid < st) { ss[tid] += ss[tid + st]; sq[tid] += sq[tid + st]; }
    __syncthreads();
  }
  if (tid == 0) {
    float m = ss[0] / (float)tot;
    float var = sq[0] / (float)tot - m * m;
    mean[c] = m;
    istd[c] = rsqrtf(var + 1e-5f);
  }
}

// act: 0 none, 1 relu, 2 gelu;  res (nullable) added before activation
__global__ void k_bn_apply(const float* __restrict__ x, const float* __restrict__ mean,
                           const float* __restrict__ istd, const float* __restrict__ g,
                           const float* __restrict__ bta, const float* __restrict__ res,
                           float* __restrict__ out, int C, int Lc, int act) {
  int idx = blockIdx.x * blockDim.x + threadIdx.x;
  int tot = B_SZ * C * Lc;
  if (idx >= tot) return;
  int c = (idx / Lc) % C;
  float v = (x[idx] - mean[c]) * istd[c] * g[c] + bta[c];
  if (res) v += res[idx];
  if (act == 1) v = fmaxf(v, 0.f);
  else if (act == 2) v = gelu_(v);
  out[idx] = v;
}

// ---------------------------------------------------------------------------
// 4. MaxPool k3 s2 p1
// ---------------------------------------------------------------------------
__global__ void k_maxpool(const float* __restrict__ in, float* __restrict__ out,
                          int C, int Lin, int Lout) {
  int idx = blockIdx.x * blockDim.x + threadIdx.x;
  int tot = B_SZ * C * Lout;
  if (idx >= tot) return;
  int o = idx % Lout;
  int c = (idx / Lout) % C;
  int b = idx / (Lout * C);
  const float* ip = in + ((size_t)b * C + c) * Lin;
  float m = -3.4e38f;
#pragma unroll
  for (int i = 0; i < 3; ++i) {
    int li = 2 * o + i - 1;
    if (li >= 0 && li < Lin) m = fmaxf(m, ip[li]);
  }
  out[idx] = m;
}

// ---------------------------------------------------------------------------
// 5. Channel-offset copy (concat) and NCH -> token-row transpose
// ---------------------------------------------------------------------------
__global__ void k_copy_chan(const float* __restrict__ src, float* __restrict__ dst,
                            int Cs, int Lc, int Cd, int coff) {
  int idx = blockIdx.x * blockDim.x + threadIdx.x;
  int tot = B_SZ * Cs * Lc;
  if (idx >= tot) return;
  int l = idx % Lc;
  int c = (idx / Lc) % Cs;
  int b = idx / (Lc * Cs);
  dst[((size_t)b * Cd + coff + c) * Lc + l] = src[idx];
}

__global__ void k_nch_to_rows(const float* __restrict__ src, float* __restrict__ dst) {
  int idx = blockIdx.x * blockDim.x + threadIdx.x;
  if (idx >= MP_ROWS * DM) return;
  int r = idx / DM, c = idx % DM;
  float v = 0.f;
  if (r < M_ROWS) {
    int b = r / L_TOK, l = r % L_TOK;
    v = src[((size_t)b * DM + c) * L_TOK + l];
  }
  dst[idx] = v;
}

// ---------------------------------------------------------------------------
// 6. LayerNorm over last dim (one thread per row)
// ---------------------------------------------------------------------------
__global__ void k_ln_rows(const float* __restrict__ x, const float* __restrict__ g,
                          const float* __restrict__ bta, float* __restrict__ out,
                          int M, int C) {
  int r = blockIdx.x * blockDim.x + threadIdx.x;
  if (r >= M) return;
  const float* xr = x + (size_t)r * C;
  float m = 0.f;
  for (int c = 0; c < C; ++c) m += xr[c];
  m /= (float)C;
  float v = 0.f;
  for (int c = 0; c < C; ++c) { float d = xr[c] - m; v += d * d; }
  v /= (float)C;
  float is = rsqrtf(v + 1e-5f);
  float* orow = out + (size_t)r * C;
  for (int c = 0; c < C; ++c) orow[c] = (xr[c] - m) * is * g[c] + bta[c];
}

// ---------------------------------------------------------------------------
// 7. f32 -> bf16 staging casts (row-padded / K-padded / optional per-batch flip)
// ---------------------------------------------------------------------------
__global__ void k_cast_rows(const float* __restrict__ src, __bf16* __restrict__ dst,
                            int M, int K, int Kp, int srcLd, int coff, int flip) {
  int idx = blockIdx.x * blockDim.x + threadIdx.x;
  if (idx >= MP_ROWS * Kp) return;
  int r = idx / Kp, k = idx % Kp;
  float v = 0.f;
  if (r < M && k < K) {
    int rr = r;
    if (flip) { int b = r / L_TOK, l = r % L_TOK; rr = b * L_TOK + (L_TOK - 1 - l); }
    v = src[(size_t)rr * srcLd + coff + k];
  }
  dst[idx] = f2bf(v);
}

__global__ void k_cast_w(const float* __restrict__ src, __bf16* __restrict__ dst,
                         int N, int K, int Np, int Kp) {
  int idx = blockIdx.x * blockDim.x + threadIdx.x;
  if (idx >= Np * Kp) return;
  int n = idx / Kp, k = idx % Kp;
  float v = (n < N && k < K) ? src[(size_t)n * K + k] : 0.f;
  dst[idx] = f2bf(v);
}

// ---------------------------------------------------------------------------
// 8. WMMA GEMM:  C[Mp x Np] = A[Mp x Kp] * W[Np x Kp]^T   (bf16 in, f32 acc)
//    One wave per block computes a 16x64 macro-tile (4 accumulators, A reused
//    4x). Np is guaranteed a multiple of 64 by the caller (all staging buffers
//    are padded), so the inner loop is branch-free: 5 clause-able b128 load
//    pairs followed by 4 back-to-back WMMAs.
//    Fragment mapping used here (contiguous K order, same for A and B):
//      A 16x32 : M = lane%16, 16 contiguous K at k0 + (lane>=16 ? 16 : 0)
//      B 32x16 : N = lane%16, 16 contiguous K at k0 + (lane>=16 ? 16 : 0)
//      D f32   : elem r -> (M = r + (lane>=16 ? 8 : 0), N = lane%16)
// ---------------------------------------------------------------------------
__global__ void k_wmma_gemm(const __bf16* __restrict__ A, const __bf16* __restrict__ W,
                            float* __restrict__ C, int Kp, int Np) {
  const int lane  = threadIdx.x;          // 0..31 (wave32)
  const int hl    = lane >> 4;            // half-select
  const int l16   = lane & 15;
  const int mtile = blockIdx.x * 16;
  const int nbase = blockIdx.y * 64;
  union FragU { v16bf v; v8bf h[2]; };

  v8f acc0 = {}, acc1 = {}, acc2 = {}, acc3 = {};
  const __bf16* ap  = A + (size_t)(mtile + l16) * Kp + hl * 16;
  const __bf16* bp0 = W + (size_t)(nbase + l16) * Kp + hl * 16;
  const __bf16* bp1 = bp0 + (size_t)16 * Kp;
  const __bf16* bp2 = bp0 + (size_t)32 * Kp;
  const __bf16* bp3 = bp0 + (size_t)48 * Kp;

  for (int k0 = 0; k0 < Kp; k0 += 32) {
    FragU af, f0, f1, f2, f3;
    af.h[0] = *(const v8bf*)(ap);       af.h[1] = *(const v8bf*)(ap + 8);
    f0.h[0] = *(const v8bf*)(bp0);      f0.h[1] = *(const v8bf*)(bp0 + 8);
    f1.h[0] = *(const v8bf*)(bp1);      f1.h[1] = *(const v8bf*)(bp1 + 8);
    f2.h[0] = *(const v8bf*)(bp2);      f2.h[1] = *(const v8bf*)(bp2 + 8);
    f3.h[0] = *(const v8bf*)(bp3);      f3.h[1] = *(const v8bf*)(bp3 + 8);
    acc0 = __builtin_amdgcn_wmma_f32_16x16x32_bf16(false, af.v, false, f0.v, (short)0, acc0, false, false);
    acc1 = __builtin_amdgcn_wmma_f32_16x16x32_bf16(false, af.v, false, f1.v, (short)0, acc1, false, false);
    acc2 = __builtin_amdgcn_wmma_f32_16x16x32_bf16(false, af.v, false, f2.v, (short)0, acc2, false, false);
    acc3 = __builtin_amdgcn_wmma_f32_16x16x32_bf16(false, af.v, false, f3.v, (short)0, acc3, false, false);
    ap += 32; bp0 += 32; bp1 += 32; bp2 += 32; bp3 += 32;
  }

#pragma unroll
  for (int t = 0; t < 4; ++t) {
    v8f a = (t == 0) ? acc0 : (t == 1) ? acc1 : (t == 2) ? acc2 : acc3;
    int n = nbase + t * 16 + l16;
    float* cp = C + (size_t)(mtile + hl * 8) * Np + n;
#pragma unroll
    for (int r = 0; r < 8; ++r) cp[(size_t)r * Np] = a[r];
  }
}

// ---------------------------------------------------------------------------
// 9. Mamba inner kernels
// ---------------------------------------------------------------------------
// causal depthwise conv k4 (pad 3 left) over xz[:, 0:256] + SiLU -> xi
__global__ void k_mamba_conv(const float* __restrict__ xz, const float* __restrict__ cw,
                             const float* __restrict__ cb, float* __restrict__ xi) {
  int idx = blockIdx.x * blockDim.x + threadIdx.x;
  if (idx >= M_ROWS * DI) return;
  int d = idx % DI, r = idx / DI;
  int b = r / L_TOK, l = r % L_TOK;
  float acc = cb[d];
#pragma unroll
  for (int i = 0; i < 4; ++i) {
    int li = l + i - 3;
    if (li >= 0) acc += cw[d * 4 + i] * xz[((size_t)(b * L_TOK + li)) * XZ_LD + d];
  }
  xi[idx] = acc * sigmoidf_(acc);
}

__global__ void k_softplus(float* __restrict__ dt, const float* __restrict__ dt_b) {
  int idx = blockIdx.x * blockDim.x + threadIdx.x;
  if (idx >= M_ROWS * DI) return;
  float v = dt[idx] + dt_b[idx % DI];
  dt[idx] = (v > 20.f) ? v : log1pf(__expf(v));
}

__global__ void k_aneg(const float* __restrict__ A_log, float* __restrict__ An) {
  int idx = blockIdx.x * blockDim.x + threadIdx.x;
  if (idx < DI * DS) An[idx] = -__expf(A_log[idx]);
}

// selective scan: one thread per (b, d_inner) recurrence, h[16] in registers
__global__ void k_scan(const float* __restrict__ delta, const float* __restrict__ dbl,
                       const float* __restrict__ xi, const float* __restrict__ An,
                       const float* __restrict__ Dv, float* __restrict__ y) {
  int idx = blockIdx.x * blockDim.x + threadIdx.x;
  if (idx >= B_SZ * DI) return;
  int b = idx / DI, d = idx % DI;
  float h[DS];
  float a[DS];
#pragma unroll
  for (int s = 0; s < DS; ++s) { h[s] = 0.f; a[s] = An[d * DS + s]; }
  float Dd = Dv[d];
  for (int t = 0; t < L_TOK; ++t) {
    size_t r = (size_t)(b * L_TOK + t);
    if (t + 1 < L_TOK)
      __builtin_prefetch((const void*)(dbl + (r + 1) * DBL_LD), 0, 1);
    float dl = delta[r * DI + d];
    float xv = xi[r * DI + d];
    const float* Bc = dbl + r * DBL_LD + DTR;        // B: cols 8..23
    const float* Cc = dbl + r * DBL_LD + DTR + DS;   // C: cols 24..39
    float yv = 0.f;
#pragma unroll
    for (int s = 0; s < DS; ++s) {
      float dA = __expf(dl * a[s]);
      h[s] = dA * h[s] + dl * Bc[s] * xv;
      yv += h[s] * Cc[s];
    }
    y[r * DI + d] = yv + xv * Dd;
  }
}

__global__ void k_gate(float* __restrict__ y, const float* __restrict__ xz) {
  int idx = blockIdx.x * blockDim.x + threadIdx.x;
  if (idx >= M_ROWS * DI) return;
  int d = idx % DI, r = idx / DI;
  float z = xz[(size_t)r * XZ_LD + DI + d];
  y[idx] *= z * sigmoidf_(z);
}

__global__ void k_accum(float* __restrict__ dst, const float* __restrict__ src, int flip) {
  int idx = blockIdx.x * blockDim.x + threadIdx.x;
  if (idx >= M_ROWS * DM) return;
  int r = idx / DM, c = idx % DM;
  int rr = r;
  if (flip) { int b = r / L_TOK, l = r % L_TOK; rr = b * L_TOK + (L_TOK - 1 - l); }
  dst[idx] += src[(size_t)rr * DM + c];
}

// ---------------------------------------------------------------------------
// 10. Head: mean pool + MLP classifier
// ---------------------------------------------------------------------------
__global__ void k_pool(const float* __restrict__ x, float* __restrict__ pooled) {
  int idx = blockIdx.x * blockDim.x + threadIdx.x;
  if (idx >= B_SZ * DM) return;
  int b = idx / DM, c = idx % DM;
  float s = 0.f;
  for (int l = 0; l < L_TOK; ++l) s += x[((size_t)(b * L_TOK + l)) * DM + c];
  pooled[idx] = s / (float)L_TOK;
}

__global__ void k_cls1(const float* __restrict__ pooled, const float* __restrict__ w1,
                       const float* __restrict__ b1, float* __restrict__ h) {
  int idx = blockIdx.x * blockDim.x + threadIdx.x;
  if (idx >= B_SZ * 64) return;
  int b = idx / 64, j = idx % 64;
  float s = b1[j];
  for (int c = 0; c < DM; ++c) s += pooled[b * DM + c] * w1[j * DM + c];
  h[idx] = gelu_(s);
}

__global__ void k_cls2(const float* __restrict__ h, const float* __restrict__ w2,
                       const float* __restrict__ b2, float* __restrict__ out) {
  int idx = blockIdx.x * blockDim.x + threadIdx.x;
  if (idx >= B_SZ * 2) return;
  int b = idx / 2, j = idx % 2;
  float s = b2[j];
  for (int c = 0; c < 64; ++c) s += h[b * 64 + c] * w2[j * 64 + c];
  out[idx] = s;
}

// ---------------------------------------------------------------------------
// Host launcher
// ---------------------------------------------------------------------------
struct MPr {
  const float *A_log, *D, *conv_b, *conv_w, *dt_b, *dt_w, *in_w, *out_w, *x_w;
};

extern "C" void kernel_launch(void* const* d_in, const int* in_sizes, int n_in,
                              void* d_out, int out_size, void* d_ws, size_t ws_size,
                              hipStream_t stream) {
  (void)in_sizes; (void)n_in; (void)out_size; (void)ws_size;
  auto PF = [&](int i) { return (const float*)d_in[i]; };
  // Param indices assume JAX pytree (alphabetical dict-key) flatten order:
  // 0:x; blocks[0]{bwd:1-9, fwd:10-18, ln_b:19, ln_g:20}; blocks[1]{21-29,30-38,39,40};
  // 41:cls_b1 42:cls_b2 43:cls_w1 44:cls_w2 45:fln_b 46:fln_g;
  // high:47-76  low:77-106 (l1[0]:+0..5  l1[1]:+6..11  l2[0]:+12..20  l2[1]:+21..26
  //                         stem_b:+27 stem_g:+28 stem_w:+29);
  // 107:proj_b 108:proj_g 109:proj_w
  auto mp = [&](int b) {
    return MPr{PF(b), PF(b + 1), PF(b + 2), PF(b + 3), PF(b + 4),
               PF(b + 5), PF(b + 6), PF(b + 7), PF(b + 8)};
  };

  // ---- scratch carve-out ----
  char* wsb = (char*)d_ws;
  size_t off = 0;
  auto alloc = [&](size_t elems, size_t esz) -> void* {
    off = (off + 255) & ~(size_t)255;
    void* p = wsb + off;
    off += elems * esz;
    return p;
  };
  const size_t SB = (size_t)B_SZ * 32 * L_STEM;  // largest NCH intermediate
  float* yl   = (float*)alloc((size_t)B_SZ * L_DWT, 4);
  float* yh   = (float*)alloc((size_t)B_SZ * L_DWT, 4);
  float* t1   = (float*)alloc(SB, 4);
  float* t2   = (float*)alloc(SB, 4);
  float* t3   = (float*)alloc(SB, 4);
  float* t4   = (float*)alloc(SB, 4);
  float* s0   = (float*)alloc(SB, 4);
  float* catb = (float*)alloc((size_t)B_SZ * DM * L_TOK, 4);
  float* prj  = (float*)alloc((size_t)B_SZ * DM * L_TOK, 4);
  float* tok  = (float*)alloc((size_t)MP_ROWS * DM, 4);
  float* xn   = (float*)alloc((size_t)MP_ROWS * DM, 4);
  float* statm = (float*)alloc(DM, 4);
  float* stati = (float*)alloc(DM, 4);
  __bf16* xnbf = (__bf16*)alloc((size_t)MP_ROWS * DM, 2);
  __bf16* xibf = (__bf16*)alloc((size_t)MP_ROWS * DI, 2);
  __bf16* dtbf = (__bf16*)alloc((size_t)MP_ROWS * 32, 2);
  __bf16* ybf  = (__bf16*)alloc((size_t)MP_ROWS * DI, 2);
  __bf16* wbf  = (__bf16*)alloc((size_t)512 * 128, 2);
  float* xz    = (float*)alloc((size_t)MP_ROWS * XZ_LD, 4);
  float* xi    = (float*)alloc((size_t)MP_ROWS * DI, 4);
  float* dbl   = (float*)alloc((size_t)MP_ROWS * DBL_LD, 4);
  float* delta = (float*)alloc((size_t)MP_ROWS * DI, 4);
  float* aneg  = (float*)alloc((size_t)DI * DS, 4);
  float* yv    = (float*)alloc((size_t)MP_ROWS * DI, 4);
  float* mout  = (float*)alloc((size_t)MP_ROWS * DM, 4);
  float* pooled = (float*)alloc(B_SZ * DM, 4);
  float* hcls   = (float*)alloc(B_SZ * 64, 4);

  auto gemm = [&](const __bf16* A, const __bf16* W, float* C, int Kp, int Np) {
    // Np must be a multiple of 64 (all buffers are padded accordingly)
    dim3 grid(MP_ROWS / 16, (unsigned)(Np / 64));
    k_wmma_gemm<<<grid, 32, 0, stream>>>(A, W, C, Kp, Np);
  };

  auto conv_bn = [&](const float* in, const float* wgt, const float* g, const float* b,
                     float* conv_out, float* out, const float* res,
                     int Cin, int Cout, int Lin, int Lout, int K, int S, int P, int act) {
    int tot = B_SZ * Cout * Lout;
    k_conv1d<<<GE(tot), 256, 0, stream>>>(in, wgt, conv_out, Cin, Cout, Lin, Lout, K, S, P);
    k_bn_stats<<<(unsigned)Cout, 256, 0, stream>>>(conv_out, statm, stati, Cout, Lout);
    k_bn_apply<<<GE(tot), 256, 0, stream>>>(conv_out, statm, stati, g, b, res, out, Cout, Lout, act);
  };

  auto basic = [&](float* inbuf, float* outbuf, int pb, int Cin, int Cout,
                   int Lin, int Lout, int S, bool has_ws) {
    const float *b1, *b2, *bs = nullptr, *g1, *g2, *gs = nullptr, *w1, *w2, *wsm = nullptr;
    if (has_ws) { b1 = PF(pb); b2 = PF(pb + 1); bs = PF(pb + 2); g1 = PF(pb + 3);
                  g2 = PF(pb + 4); gs = PF(pb + 5); w1 = PF(pb + 6); w2 = PF(pb + 7);
                  wsm = PF(pb + 8); }
    else        { b1 = PF(pb); b2 = PF(pb + 1); g1 = PF(pb + 2); g2 = PF(pb + 3);
                  w1 = PF(pb + 4); w2 = PF(pb + 5); }
    const float* res;
    if (has_ws) {
      conv_bn(inbuf, wsm, gs, bs, t3, t4, nullptr, Cin, Cout, Lin, Lout, 1, S, 0, 0);
      res = t4;
    } else res = inbuf;
    conv_bn(inbuf, w1, g1, b1, t1, t2, nullptr, Cin, Cout, Lin, Lout, 3, S, 1, 1);
    conv_bn(t2, w2, g2, b2, t1, outbuf, res, Cout, Cout, Lout, Lout, 3, 1, 1, 1);
  };

  auto run_stream = [&](const float* ydwt, int pb, int coff) {
    conv_bn(ydwt, PF(pb + 29), PF(pb + 28), PF(pb + 27), t1, t2, nullptr,
            1, 32, L_DWT, L_STEM, 7, 2, 3, 1);
    k_maxpool<<<GE(B_SZ * 32 * L_POOL), 256, 0, stream>>>(t2, s0, 32, L_STEM, L_POOL);
    basic(s0, s0, pb + 0, 32, 32, L_POOL, L_POOL, 1, false);
    basic(s0, s0, pb + 6, 32, 32, L_POOL, L_POOL, 1, false);
    basic(s0, s0, pb + 12, 32, 64, L_POOL, L_TOK, 2, true);
    basic(s0, s0, pb + 21, 64, 64, L_TOK, L_TOK, 1, false);
    k_copy_chan<<<GE(B_SZ * 64 * L_TOK), 256, 0, stream>>>(s0, catb, 64, L_TOK, DM, coff);
  };

  auto run_mamba = [&](const MPr& m, int flip) {
    // in_proj: xz = xn @ in_w^T  (WMMA bf16)
    k_cast_rows<<<GE(MP_ROWS * DM), 256, 0, stream>>>(xn, xnbf, M_ROWS, DM, DM, DM, 0, flip);
    k_cast_w<<<GE(512 * 128), 256, 0, stream>>>(m.in_w, wbf, 512, 128, 512, 128);
    gemm(xnbf, wbf, xz, 128, 512);
    // depthwise causal conv + silu
    k_mamba_conv<<<GE(M_ROWS * DI), 256, 0, stream>>>(xz, m.conv_w, m.conv_b, xi);
    // x_proj: dbl = xi @ x_w^T  (N=40 padded to 64)
    k_cast_rows<<<GE(MP_ROWS * DI), 256, 0, stream>>>(xi, xibf, M_ROWS, DI, DI, DI, 0, 0);
    k_cast_w<<<GE(DBL_LD * 256), 256, 0, stream>>>(m.x_w, wbf, 40, 256, DBL_LD, 256);
    gemm(xibf, wbf, dbl, 256, DBL_LD);
    // dt_proj: delta_raw = dt @ dt_w^T  (K=8 padded to 32)
    k_cast_rows<<<GE(MP_ROWS * 32), 256, 0, stream>>>(dbl, dtbf, M_ROWS, DTR, 32, DBL_LD, 0, 0);
    k_cast_w<<<GE(256 * 32), 256, 0, stream>>>(m.dt_w, wbf, 256, 8, 256, 32);
    gemm(dtbf, wbf, delta, 32, 256);
    k_softplus<<<GE(M_ROWS * DI), 256, 0, stream>>>(delta, m.dt_b);
    k_aneg<<<GE(DI * DS), 256, 0, stream>>>(m.A_log, aneg);
    // selective scan + skip (D)
    k_scan<<<GE(B_SZ * DI), 256, 0, stream>>>(delta, dbl, xi, aneg, m.D, yv);
    k_gate<<<GE(M_ROWS * DI), 256, 0, stream>>>(yv, xz);
    // out_proj + residual accumulate (unflip for bwd)
    k_cast_rows<<<GE(MP_ROWS * DI), 256, 0, stream>>>(yv, ybf, M_ROWS, DI, DI, DI, 0, 0);
    k_cast_w<<<GE(128 * 256), 256, 0, stream>>>(m.out_w, wbf, 128, 256, 128, 256);
    gemm(ybf, wbf, mout, 256, 128);
    k_accum<<<GE(M_ROWS * DM), 256, 0, stream>>>(tok, mout, flip);
  };

  // ================= pipeline =================
  k_dwt<<<GE(B_SZ * L_DWT), 256, 0, stream>>>(PF(0), yl, yh);
  run_stream(yl, 77, 0);    // low stream  -> channels [0,64)
  run_stream(yh, 47, 64);   // high stream -> channels [64,128)
  // proj 1x1 conv + BN + exact GELU
  conv_bn(catb, PF(109), PF(108), PF(107), prj, prj, nullptr, DM, DM, L_TOK, L_TOK, 1, 1, 0, 2);
  k_nch_to_rows<<<GE(MP_ROWS * DM), 256, 0, stream>>>(prj, tok);

  for (int blk = 0; blk < 2; ++blk) {
    int fb  = blk ? 30 : 10;
    int bb  = blk ? 21 : 1;
    int lnb = blk ? 39 : 19;
    int lng = blk ? 40 : 20;
    k_ln_rows<<<GE(M_ROWS), 256, 0, stream>>>(tok, PF(lng), PF(lnb), xn, M_ROWS, DM);
    run_mamba(mp(fb), 0);
    run_mamba(mp(bb), 1);
  }

  k_ln_rows<<<GE(M_ROWS), 256, 0, stream>>>(tok, PF(46), PF(45), xn, M_ROWS, DM);
  k_pool<<<GE(B_SZ * DM), 256, 0, stream>>>(xn, pooled);
  k_cls1<<<GE(B_SZ * 64), 256, 0, stream>>>(pooled, PF(43), PF(41), hcls);
  k_cls2<<<1, 64, 0, stream>>>(hcls, PF(44), PF(42), (float*)d_out);
}
